// TemporalSelfAttentionLite_90718299226274
// MI455X (gfx1250) — matrix operand order repaired and checked by reading, not compiled, not run on hardware
//
#include <hip/hip_runtime.h>
#include <hip/hip_bf16.h>

// ---------------------------------------------------------------------------
// TemporalSelfAttentionLite for MI455X (gfx1250, wave32, WMMA bf16)
//   BS=2 NBQ=2 N=4 NQ=40000 C=256 NH=8 HD=32 NP=4 H=W=200
// Pipeline:
//   k_vproj   : v = value @ W_v^T + b_v      -> bf16 images (n,h,y,x,d)  [WMMA]
//   k_offattn : raw = q_cat @ [W_off;W_attn]^T + bias  (80000 x 192 f32) [WMMA]
//   k_sample  : softmax + bilinear gather + head-merge -> pre_out bf16
//   k_outproj : out = pre_out @ W_out^T + b_out + query (residual)       [WMMA]
// GEMM waves own 2 M-tiles x (8|6) col-tiles so every LDS B-fragment feeds
// two WMMAs. f32->bf16 packing = add-0x8000 + one v_perm_b32 per pair.
// ---------------------------------------------------------------------------

#define kNQ   40000
#define kC    256
#define kNH   8
#define kHD   32
#define kNP   4
#define kHH   200
#define kWW   200
#define kHW   (kHH * kWW)
#define kMROWS_V   (4 * kNQ)   // 160000
#define kMROWS_Q   (2 * kNQ)   // 80000
#define kKOFF 512              // q_cat dim
#define kNOUT2 192             // 128 soff + 64 attn

typedef __attribute__((ext_vector_type(16))) __bf16 v16bf;
typedef __attribute__((ext_vector_type(8)))  float  v8f;

union Frag {
  uint4        q[2];
  unsigned int u[8];
  v16bf        v;
};

// f32 -> bf16, round-half-up (differs from RNE only on exact ties).
__device__ __forceinline__ unsigned short f2bf(float f) {
  unsigned int u = __builtin_bit_cast(unsigned int, f) + 0x8000u;
  return (unsigned short)(u >> 16);
}
__device__ __forceinline__ float bf2f(unsigned short s) {
  unsigned int u = ((unsigned int)s) << 16;
  return __builtin_bit_cast(float, u);
}
// Pack bf16(a) | bf16(b)<<16 : two v_add + one v_perm_b32.
__device__ __forceinline__ unsigned int pk2(float a, float b) {
  unsigned int ua = __builtin_bit_cast(unsigned int, a) + 0x8000u;
  unsigned int ub = __builtin_bit_cast(unsigned int, b) + 0x8000u;
  // dst bytes {0,1,2,3} = {ua.b2, ua.b3, ub.b2, ub.b3}
  return __builtin_amdgcn_perm(ub, ua, 0x07060302u);
}
__device__ __forceinline__ v8f wmma_bf16(v16bf a, v16bf b, v8f c) {
  // D = A(16x32 bf16) * B(32x16 bf16) + C(16x16 f32)
  return __builtin_amdgcn_wmma_f32_16x16x32_bf16(
      false, a, false, b, (short)0, c, false, false);
}

// Build a 16x32 bf16 A-fragment from fp32 rows (documented VGPR layout:
// halves 0..7 = K koff..koff+7, halves 8..15 = K 16+koff..16+koff+7).
__device__ __forceinline__ Frag load_a_f32(const float* ap) {
  float4 f0 = *(const float4*)(ap + 0);
  float4 f1 = *(const float4*)(ap + 4);
  float4 f2 = *(const float4*)(ap + 16);
  float4 f3 = *(const float4*)(ap + 20);
  Frag fa;
  fa.u[0] = pk2(f0.x, f0.y); fa.u[1] = pk2(f0.z, f0.w);
  fa.u[2] = pk2(f1.x, f1.y); fa.u[3] = pk2(f1.z, f1.w);
  fa.u[4] = pk2(f2.x, f2.y); fa.u[5] = pk2(f2.z, f2.w);
  fa.u[6] = pk2(f3.x, f3.y); fa.u[7] = pk2(f3.z, f3.w);
  return fa;
}

// ---------------------------------------------------------------------------
// Kernel 1: v projection. 8 waves/block = 4 row-pairs x 2 col-groups.
// LDS holds W_v^T pre-swizzled into B-fragment order: [16 ct][8 ks][32 ln][16]
// ---------------------------------------------------------------------------
__global__ void __launch_bounds__(256)
k_vproj(const float* __restrict__ value, const float* __restrict__ Wv,
        const float* __restrict__ bv, unsigned short* __restrict__ vimg) {
  extern __shared__ unsigned short ldsB[];
  const int tid = threadIdx.x;
  for (int e2 = tid; e2 < 16 * 8 * 32 * 8; e2 += 256) {   // pair-packed fill
    const int e = e2 * 2;
    const int i = e & 15, ln = (e >> 4) & 31, ks = (e >> 9) & 7, ct = e >> 12;
    const int kk = ks * 32 + ((ln & 16) ? 8 : 0) + (i & 7) + ((i & 8) ? 16 : 0);    // B: lanes16-31 K+8, elems 8..15 K+16
    const int nn = ct * 16 + (ln & 15);
    const float* wp = Wv + (size_t)nn * kC + kk;          // B[k][n] = W_v[n][k]
    ((unsigned int*)ldsB)[e2] = pk2(wp[0], wp[1]);
  }
  __syncthreads();

  const int lane = tid & 31, wave = tid >> 5;
  const int rowBase = blockIdx.x * 128 + (wave & 3) * 32;
  const int ctBase  = (wave >> 2) * 8;
  const int row0 = rowBase + (lane & 15);
  const int row1 = row0 + 16;
  const int koff = (lane < 16) ? 0 : 8;                   // A K-offset per half

  v8f acc[2][8];
  #pragma unroll
  for (int t = 0; t < 2; ++t)
    #pragma unroll
    for (int c = 0; c < 8; ++c)
      #pragma unroll
      for (int j = 0; j < 8; ++j) acc[t][c][j] = 0.f;

  for (int ks = 0; ks < 8; ++ks) {
    Frag fa0 = load_a_f32(value + (size_t)row0 * kC + ks * 32 + koff);
    Frag fa1 = load_a_f32(value + (size_t)row1 * kC + ks * 32 + koff);
    #pragma unroll
    for (int c = 0; c < 8; ++c) {
      Frag fb;
      const uint4* bp = (const uint4*)
          (ldsB + ((size_t)((ctBase + c) * 8 + ks) * 32 + lane) * 16);
      fb.q[0] = bp[0]; fb.q[1] = bp[1];
      acc[0][c] = wmma_bf16(fa0.v, fb.v, acc[0][c]);
      acc[1][c] = wmma_bf16(fa1.v, fb.v, acc[1][c]);
    }
  }

  // Epilogue: scatter into per-head bf16 images (n,h,y,x,d). Row-pair tiles
  // never straddle n since NQ % 32 == 0.
  const int n = rowBase / kNQ;
  const int yxBase = rowBase % kNQ;
  #pragma unroll
  for (int t = 0; t < 2; ++t) {
    #pragma unroll
    for (int c = 0; c < 8; ++c) {
      const int col = (ctBase + c) * 16 + (lane & 15);
      const int h = col >> 5, d = col & 31;
      const float bias = bv[col];
      #pragma unroll
      for (int j = 0; j < 8; ++j) {
        const int yx = yxBase + t * 16 + ((lane < 16) ? j : 8 + j);
        const size_t idx = ((size_t)(n * kNH + h) * kHW + yx) * (size_t)kHD + d;
        vimg[idx] = f2bf(acc[t][c][j] + bias);
      }
    }
  }
}

// ---------------------------------------------------------------------------
// Kernel 2: sampling offsets + attention logits GEMM (K=512, Nout=192).
// 8 waves = 4 row-pairs x 2 col-groups of 6 tiles.
// ---------------------------------------------------------------------------
__global__ void __launch_bounds__(256)
k_offattn(const float* __restrict__ value, const float* __restrict__ query,
          const float* __restrict__ Woff, const float* __restrict__ boff,
          const float* __restrict__ Wattn, const float* __restrict__ battn,
          float* __restrict__ raw) {
  extern __shared__ unsigned short ldsB[];   // [12 ct][16 ks][32 ln][16]
  const int tid = threadIdx.x;
  for (int e2 = tid; e2 < 12 * 16 * 32 * 8; e2 += 256) {
    const int e = e2 * 2;
    const int i = e & 15, ln = (e >> 4) & 31, ks = (e >> 9) & 15, ct = e >> 13;
    const int kk = ks * 32 + ((ln & 16) ? 8 : 0) + (i & 7) + ((i & 8) ? 16 : 0);
    const int nn = ct * 16 + (ln & 15);
    const float* wp = (nn < 128) ? (Woff + (size_t)nn * kKOFF + kk)
                                 : (Wattn + (size_t)(nn - 128) * kKOFF + kk);
    ((unsigned int*)ldsB)[e2] = pk2(wp[0], wp[1]);
  }
  __syncthreads();

  const int lane = tid & 31, wave = tid >> 5;
  const int rowBase = blockIdx.x * 128 + (wave & 3) * 32;
  const int ctBase  = (wave >> 2) * 6;
  const int row0 = rowBase + (lane & 15);
  const int row1 = row0 + 16;
  const int koff = (lane < 16) ? 0 : 8;

  v8f acc[2][6];
  #pragma unroll
  for (int t = 0; t < 2; ++t)
    #pragma unroll
    for (int c = 0; c < 6; ++c)
      #pragma unroll
      for (int j = 0; j < 8; ++j) acc[t][c][j] = 0.f;

  for (int ks = 0; ks < 16; ++ks) {
    const int kb = ks * 32;
    // q_cat = [value[:2] (256) | query (256)]
    const float* base0 = (kb < kC) ? (value + (size_t)row0 * kC + kb + koff)
                                   : (query + (size_t)row0 * kC + (kb - kC) + koff);
    const float* base1 = (kb < kC) ? (value + (size_t)row1 * kC + kb + koff)
                                   : (query + (size_t)row1 * kC + (kb - kC) + koff);
    Frag fa0 = load_a_f32(base0);
    Frag fa1 = load_a_f32(base1);
    #pragma unroll
    for (int c = 0; c < 6; ++c) {
      Frag fb;
      const uint4* bp = (const uint4*)
          (ldsB + ((size_t)((ctBase + c) * 16 + ks) * 32 + lane) * 16);
      fb.q[0] = bp[0]; fb.q[1] = bp[1];
      acc[0][c] = wmma_bf16(fa0.v, fb.v, acc[0][c]);
      acc[1][c] = wmma_bf16(fa1.v, fb.v, acc[1][c]);
    }
  }

  #pragma unroll
  for (int t = 0; t < 2; ++t) {
    #pragma unroll
    for (int c = 0; c < 6; ++c) {
      const int col = (ctBase + c) * 16 + (lane & 15);
      const float bias = (col < 128) ? boff[col] : battn[col - 128];
      #pragma unroll
      for (int j = 0; j < 8; ++j) {
        const int m = rowBase + t * 16 + ((lane < 16) ? j : 8 + j);
        raw[(size_t)m * kNOUT2 + col] = acc[t][c][j] + bias;
      }
    }
  }
}

// ---------------------------------------------------------------------------
// Kernel 3: softmax + bilinear sampling + NBQ mean. One wave per (b,q);
// lane = channel d. Lanes g=0..15 own (h,nbq) groups; coords/weights are
// wave-broadcast via shuffles so corner loads are uniform 64B bursts from
// the L2-resident bf16 value images.
// ---------------------------------------------------------------------------
__global__ void __launch_bounds__(256)
k_sample(const float* __restrict__ raw, const float* __restrict__ refp,
         const unsigned short* __restrict__ vimg,
         unsigned short* __restrict__ preout) {
  const int lane = threadIdx.x & 31;
  const int wave = threadIdx.x >> 5;
  const int r = blockIdx.x * 8 + wave;          // 0 .. 79999
  const int b = r / kNQ;
  const int q = r - b * kNQ;

  // Per-group (h,nbq) setup; lanes 16..31 duplicate lanes 0..15.
  const int g   = lane & 15;                    // g = h*2 + nbq
  const int ngr = b * 2 + (g & 1);              // value-batch for this group
  const float* sp = raw + (size_t)r * kNOUT2 + g * 8;
  const float* tp = raw + (size_t)r * kNOUT2 + 128 + g * 4;
  float offx[kNP], offy[kNP], att[kNP];
  #pragma unroll
  for (int p = 0; p < kNP; ++p) { offx[p] = sp[p * 2]; offy[p] = sp[p * 2 + 1]; }
  #pragma unroll
  for (int p = 0; p < kNP; ++p) att[p] = tp[p];
  float mx = fmaxf(fmaxf(att[0], att[1]), fmaxf(att[2], att[3]));
  float s = 0.f;
  #pragma unroll
  for (int p = 0; p < kNP; ++p) { att[p] = __expf(att[p] - mx); s += att[p]; }
  const float inv = 1.f / s;
  #pragma unroll
  for (int p = 0; p < kNP; ++p) att[p] *= inv;

  const float rx = refp[((size_t)ngr * kNQ + q) * 2 + 0];
  const float ry = refp[((size_t)ngr * kNQ + q) * 2 + 1];
  float sx[kNP], sy[kNP];
  #pragma unroll
  for (int p = 0; p < kNP; ++p) {
    // x = ((2*loc-1)+1)*W/2 - 0.5 == ref*W + off - 0.5
    sx[p] = rx * (float)kWW + offx[p] - 0.5f;
    sy[p] = ry * (float)kHH + offy[p] - 0.5f;
  }

  #pragma unroll
  for (int h = 0; h < kNH; ++h) {
    float accum = 0.f;
    #pragma unroll
    for (int nb = 0; nb < 2; ++nb) {
      const int gsrc = h * 2 + nb;
      const size_t ibase = (size_t)((b * 2 + nb) * kNH + h) * ((size_t)kHW * kHD);
      #pragma unroll
      for (int p = 0; p < kNP; ++p) {
        const float X = __shfl(sx[p], gsrc, 32);
        const float Y = __shfl(sy[p], gsrc, 32);
        const float W = __shfl(att[p], gsrc, 32);
        const float x0f = floorf(X), y0f = floorf(Y);
        const int ix = (int)x0f, iy = (int)y0f;
        const float fx = X - x0f, fy = Y - y0f;
        const bool xv0 = (ix >= 0) && (ix < kWW);
        const bool xv1 = (ix >= -1) && (ix < kWW - 1);
        const bool yv0 = (iy >= 0) && (iy < kHH);
        const bool yv1 = (iy >= -1) && (iy < kHH - 1);
        float v00 = 0.f, v01 = 0.f, v10 = 0.f, v11 = 0.f;
        if (xv0 && yv0) v00 = bf2f(vimg[ibase + (size_t)(iy * kWW + ix) * kHD + lane]);
        if (xv1 && yv0) v01 = bf2f(vimg[ibase + (size_t)(iy * kWW + ix + 1) * kHD + lane]);
        if (xv0 && yv1) v10 = bf2f(vimg[ibase + (size_t)((iy + 1) * kWW + ix) * kHD + lane]);
        if (xv1 && yv1) v11 = bf2f(vimg[ibase + (size_t)((iy + 1) * kWW + ix + 1) * kHD + lane]);
        const float bil = v00 * (1.f - fx) * (1.f - fy) + v01 * fx * (1.f - fy)
                        + v10 * (1.f - fx) * fy         + v11 * fx * fy;
        accum += W * bil;
      }
    }
    preout[(size_t)r * kC + h * kHD + lane] = f2bf(0.5f * accum);  // mean over NBQ
  }
}

// ---------------------------------------------------------------------------
// Kernel 4: output projection + bias + residual. A is bf16 pre_out.
// ---------------------------------------------------------------------------
__global__ void __launch_bounds__(256)
k_outproj(const unsigned short* __restrict__ preout, const float* __restrict__ Wout,
          const float* __restrict__ bout, const float* __restrict__ query,
          float* __restrict__ out) {
  extern __shared__ unsigned short ldsB[];   // [16 ct][8 ks][32 ln][16]
  const int tid = threadIdx.x;
  for (int e2 = tid; e2 < 16 * 8 * 32 * 8; e2 += 256) {
    const int e = e2 * 2;
    const int i = e & 15, ln = (e >> 4) & 31, ks = (e >> 9) & 7, ct = e >> 12;
    const int kk = ks * 32 + ((ln & 16) ? 8 : 0) + (i & 7) + ((i & 8) ? 16 : 0);
    const int nn = ct * 16 + (ln & 15);
    const float* wp = Wout + (size_t)nn * kC + kk;
    ((unsigned int*)ldsB)[e2] = pk2(wp[0], wp[1]);
  }
  __syncthreads();

  const int lane = tid & 31, wave = tid >> 5;
  const int rowBase = blockIdx.x * 128 + (wave & 3) * 32;
  const int ctBase  = (wave >> 2) * 8;
  const int row0 = rowBase + (lane & 15);
  const int row1 = row0 + 16;
  const int koff = (lane < 16) ? 0 : 8;

  v8f acc[2][8];
  #pragma unroll
  for (int t = 0; t < 2; ++t)
    #pragma unroll
    for (int c = 0; c < 8; ++c)
      #pragma unroll
      for (int j = 0; j < 8; ++j) acc[t][c][j] = 0.f;

  for (int ks = 0; ks < 8; ++ks) {
    const unsigned short* ap0 = preout + (size_t)row0 * kC + ks * 32 + koff;
    const unsigned short* ap1 = preout + (size_t)row1 * kC + ks * 32 + koff;
    Frag fa0, fa1;
    fa0.q[0] = *(const uint4*)(ap0);        // K koff+0..7  (8 bf16 = 16B)
    fa0.q[1] = *(const uint4*)(ap0 + 16);   // K koff+16..23
    fa1.q[0] = *(const uint4*)(ap1);
    fa1.q[1] = *(const uint4*)(ap1 + 16);
    #pragma unroll
    for (int c = 0; c < 8; ++c) {
      Frag fb;
      const uint4* bp = (const uint4*)
          (ldsB + ((size_t)((ctBase + c) * 8 + ks) * 32 + lane) * 16);
      fb.q[0] = bp[0]; fb.q[1] = bp[1];
      acc[0][c] = wmma_bf16(fa0.v, fb.v, acc[0][c]);
      acc[1][c] = wmma_bf16(fa1.v, fb.v, acc[1][c]);
    }
  }

  #pragma unroll
  for (int t = 0; t < 2; ++t) {
    #pragma unroll
    for (int c = 0; c < 8; ++c) {
      const int col = (ctBase + c) * 16 + (lane & 15);
      const float bias = bout[col];
      #pragma unroll
      for (int j = 0; j < 8; ++j) {
        const int m = rowBase + t * 16 + ((lane < 16) ? j : 8 + j);
        out[(size_t)m * kC + col] = acc[t][c][j] + bias + query[(size_t)m * kC + col];
      }
    }
  }
}

// ---------------------------------------------------------------------------
// Host entry
// ---------------------------------------------------------------------------
extern "C" void kernel_launch(void* const* d_in, const int* in_sizes, int n_in,
                              void* d_out, int out_size, void* d_ws, size_t ws_size,
                              hipStream_t stream) {
  const float* query = (const float*)d_in[0];
  const float* value = (const float*)d_in[1];
  const float* refp  = (const float*)d_in[2];
  const float* Wv    = (const float*)d_in[3];
  const float* bv    = (const float*)d_in[4];
  const float* Woff  = (const float*)d_in[5];
  const float* boff  = (const float*)d_in[6];
  const float* Wattn = (const float*)d_in[7];
  const float* battn = (const float*)d_in[8];
  const float* Wout  = (const float*)d_in[9];
  const float* bout  = (const float*)d_in[10];
  float* out = (float*)d_out;

  // Workspace layout (bytes):
  //   vimg   bf16  4*8*40000*32          = 81,920,000
  //   raw    f32   80000*192             = 61,440,000
  //   preout bf16  80000*256             = 40,960,000   total 184,320,000
  char* ws = (char*)d_ws;
  unsigned short* vimg   = (unsigned short*)(ws);
  float*          raw    = (float*)(ws + 81920000);
  unsigned short* preout = (unsigned short*)(ws + 81920000 + 61440000);

  const int LDS_256 = 16 * 8 * 32 * 16 * 2;   // 131072 B
  const int LDS_192 = 12 * 16 * 32 * 16 * 2;  // 196608 B
  (void)hipFuncSetAttribute((const void*)k_vproj,
      hipFuncAttributeMaxDynamicSharedMemorySize, LDS_256);
  (void)hipFuncSetAttribute((const void*)k_offattn,
      hipFuncAttributeMaxDynamicSharedMemorySize, LDS_192);
  (void)hipFuncSetAttribute((const void*)k_outproj,
      hipFuncAttributeMaxDynamicSharedMemorySize, LDS_256);

  k_vproj  <<<kMROWS_V / 128, 256, LDS_256, stream>>>(value, Wv, bv, vimg);
  k_offattn<<<kMROWS_Q / 128, 256, LDS_192, stream>>>(value, query, Woff, boff,
                                                      Wattn, battn, raw);
  k_sample <<<kMROWS_Q / 8,   256, 0,       stream>>>(raw, refp, vimg, preout);
  k_outproj<<<kMROWS_Q / 128, 256, LDS_256, stream>>>(preout, Wout, bout, query, out);

  (void)in_sizes; (void)n_in; (void)out_size; (void)ws_size;
}
